// ChannelInteractionModule_64080912056663
// MI455X (gfx1250) — matrix operand order repaired and checked by reading, not compiled, not run on hardware
//
#include <hip/hip_runtime.h>
#include <hip/hip_bf16.h>

// Problem constants (reference): B=32, C=2048, spatial 7x7
#define BATCH 32
#define CH    2048
#define SP    49
#define KPAD  64   // K=49 padded to 64 for bf16 WMMA (16x16x32)

typedef __attribute__((ext_vector_type(16))) __bf16 v16bf;
typedef __attribute__((ext_vector_type(8)))  __bf16 v8bf;
typedef __attribute__((ext_vector_type(8)))  float  v8f;

struct Q2 { uint4 lo, hi; };   // 32B container for a v16bf fragment
struct F8 { uint4 lo, hi; };   // 32B container for a v8f accumulator

// A fragment (16x32 bf16): lane holds row M=lane&15; kh=lane>>4.
// Elements 0..7  = K in [kh*8,    kh*8+7]    -> one b128
// Elements 8..15 = K in [16+kh*8, 16+kh*8+7] -> one b128
static __device__ __forceinline__ v16bf load_a_frag(const __bf16* row, int kb, int kh) {
  Q2 t;
  t.lo = *(const uint4*)(row + kb + kh * 8);
  t.hi = *(const uint4*)(row + kb + 16 + kh * 8);
  return __builtin_bit_cast(v16bf, t);
}

// B fragment (32x16 bf16): lane holds col N=lane&15; kh=lane>>4.
// Elements 0..7  = K in [kh*8,    kh*8+7]    -> one b128
// Elements 8..15 = K in [16+kh*8, 16+kh*8+7] -> one b128
static __device__ __forceinline__ v16bf load_b_frag(const __bf16* col, int kb, int kh) {
  Q2 t;
  t.lo = *(const uint4*)(col + kb + kh * 8);
  t.hi = *(const uint4*)(col + kb + 16 + kh * 8);
  return __builtin_bit_cast(v16bf, t);
}

// ---------------- data-layout pre-passes (f32 -> bf16) ----------------

// xp[b][c][64] = x[b][c][k] (k<49), else 0
__global__ void pad_xf(const float* __restrict__ x, __bf16* __restrict__ xp, int n) {
  int i = blockIdx.x * blockDim.x + threadIdx.x;
  if (i >= n) return;
  int k = i & 63;
  int bc = i >> 6;
  int kc = k < SP ? k : SP - 1;
  float v = x[(size_t)bc * SP + kc];
  xp[i] = (k < SP) ? (__bf16)v : (__bf16)0.0f;
}

// xt[b][n][c] = x[b][c][n]
__global__ void trans_xf(const float* __restrict__ x, __bf16* __restrict__ xt, int n) {
  int i = blockIdx.x * blockDim.x + threadIdx.x;
  if (i >= n) return;
  int c = i & (CH - 1);
  int t = i >> 11;            // CH = 2^11
  int sp = t % SP, b = t / SP;
  xt[i] = (__bf16)x[((size_t)b * CH + c) * SP + sp];
}

// wt[tap][co][ci] = conv_w[co][ci][tap]
__global__ void trans_w(const float* __restrict__ w, __bf16* __restrict__ wt, int n) {
  int i = blockIdx.x * blockDim.x + threadIdx.x;
  if (i >= n) return;
  int ci = i & (CH - 1);
  int t = i >> 11;
  int co = t & (CH - 1);
  int tap = t >> 11;
  wt[i] = (__bf16)w[((size_t)co * CH + ci) * 9 + tap];
}

// ---------------- stage 1: Gram matrix  sim = xf . xf^T ----------------
// 2x2 register tiling: wave computes 32x32. Symmetric -> store tiles
// transposed so accumulator rows are contiguous (b128 stores).
__global__ void __launch_bounds__(256)
gemm_gram(const __bf16* __restrict__ xpb, float* __restrict__ sim) {
  const int lane = threadIdx.x & 31;
  const int wid  = threadIdx.x >> 5;
  const int mr   = (blockIdx.x * 8 + wid) * 32;
  const int nc   = blockIdx.y * 32;
  const int m    = lane & 15;
  const int kh   = lane >> 4;

  const __bf16* arow[2], *bcol[2];
#pragma unroll
  for (int i = 0; i < 2; ++i) {
    arow[i] = xpb + (size_t)(mr + 16 * i + m) * KPAD;
    bcol[i] = xpb + (size_t)(nc + 16 * i + m) * KPAD;
  }

  v8f acc[2][2] = {};
#pragma unroll
  for (int kb = 0; kb < KPAD; kb += 32) {
    v16bf af[2], bf[2];
#pragma unroll
    for (int i = 0; i < 2; ++i) af[i] = load_a_frag(arow[i], kb, kh);
#pragma unroll
    for (int j = 0; j < 2; ++j) bf[j] = load_b_frag(bcol[j], kb, kh);
#pragma unroll
    for (int i = 0; i < 2; ++i)
#pragma unroll
      for (int j = 0; j < 2; ++j)
        acc[i][j] = __builtin_amdgcn_wmma_f32_16x16x32_bf16(
            false, af[i], false, bf[j], (short)0, acc[i][j], false, false);
  }
  const int col0 = lane & 15;
  const int rb   = (lane >> 4) * 8;
#pragma unroll
  for (int i = 0; i < 2; ++i)
#pragma unroll
    for (int j = 0; j < 2; ++j) {
      F8 f = __builtin_bit_cast(F8, acc[i][j]);
      float* sp = sim + (size_t)(nc + 16 * j + col0) * CH + mr + 16 * i + rb;
      *(uint4*)sp = f.lo;
      *(uint4*)(sp + 4) = f.hi;
    }
}

// ---------------- stage 2: row softmax of -sim/49 -> bf16 ----------------
__global__ void __launch_bounds__(256)
softmax_neg_rows(const float* __restrict__ sim, __bf16* __restrict__ Wsm) {
  const int row = blockIdx.x;
  const float sc = -1.0f / 49.0f;
  const float* s = sim + (size_t)row * CH;
  __shared__ float red[256];

  float mx = -3.402823466e38f;
  for (int j = threadIdx.x; j < CH; j += 256) mx = fmaxf(mx, s[j] * sc);
  red[threadIdx.x] = mx;
  __syncthreads();
  for (int o = 128; o > 0; o >>= 1) {
    if (threadIdx.x < o)
      red[threadIdx.x] = fmaxf(red[threadIdx.x], red[threadIdx.x + o]);
    __syncthreads();
  }
  mx = red[0];
  __syncthreads();

  float sum = 0.0f;
  for (int j = threadIdx.x; j < CH; j += 256) sum += __expf(s[j] * sc - mx);
  red[threadIdx.x] = sum;
  __syncthreads();
  for (int o = 128; o > 0; o >>= 1) {
    if (threadIdx.x < o) red[threadIdx.x] += red[threadIdx.x + o];
    __syncthreads();
  }
  const float inv = 1.0f / red[0];
  for (int j = threadIdx.x; j < CH; j += 256)
    Wsm[(size_t)row * CH + j] = (__bf16)(__expf(s[j] * sc - mx) * inv);
}

// ---------------- stage 3: Y^T[n][c] = sum_d W[c,d] * xt[n,d] ----------------
// 4x2 register tiling: wave computes 64 rows x 32 pixel-cols.
__global__ void __launch_bounds__(256)
gemm_mix(const __bf16* __restrict__ Wsm, const __bf16* __restrict__ xtb,
         __bf16* __restrict__ Ytb) {
  const int lane = threadIdx.x & 31;
  const int wid  = threadIdx.x >> 5;
  const int mr   = (blockIdx.x * 8 + wid) * 64;
  const int nc   = blockIdx.y * 32;
  const int m    = lane & 15;
  const int kh   = lane >> 4;

  const __bf16* arow[4], *bcol[2];
#pragma unroll
  for (int i = 0; i < 4; ++i)
    arow[i] = Wsm + (size_t)(mr + 16 * i + m) * CH;
#pragma unroll
  for (int j = 0; j < 2; ++j) {
    int colB = nc + 16 * j + m;
    int colc = colB < SP ? colB : SP - 1;   // clamp: junk cols never stored
    bcol[j] = xtb + (size_t)colc * CH;
  }

  v8f acc[4][2] = {};
  for (int kb = 0; kb < CH; kb += 32) {
    v16bf af[4], bf[2];
#pragma unroll
    for (int i = 0; i < 4; ++i) af[i] = load_a_frag(arow[i], kb, kh);
#pragma unroll
    for (int j = 0; j < 2; ++j) bf[j] = load_b_frag(bcol[j], kb, kh);
#pragma unroll
    for (int i = 0; i < 4; ++i)
#pragma unroll
      for (int j = 0; j < 2; ++j)
        acc[i][j] = __builtin_amdgcn_wmma_f32_16x16x32_bf16(
            false, af[i], false, bf[j], (short)0, acc[i][j], false, false);
  }
  const int col0 = lane & 15;
  const int rb   = (lane >> 4) * 8;
#pragma unroll
  for (int i = 0; i < 4; ++i)
#pragma unroll
    for (int j = 0; j < 2; ++j) {
      int col = nc + 16 * j + col0;
      v8bf h;
#pragma unroll
      for (int r = 0; r < 8; ++r) h[r] = (__bf16)acc[i][j][r];
      if (col < SP)  // transposed store: 8 contiguous channels -> one b128
        *(uint4*)(Ytb + (size_t)col * CH + mr + 16 * i + rb) =
            __builtin_bit_cast(uint4, h);
    }
}

// ---------------- stage 4: 3x3 conv (9 taps x GEMM) + bias + residual ------
// 4x2 register tiling: wave computes 64 c_out x 32 pixels.
__global__ void __launch_bounds__(256)
conv3x3_bias_res(const __bf16* __restrict__ Yt, const __bf16* __restrict__ wt,
                 const float* __restrict__ bias, const float* __restrict__ x,
                 float* __restrict__ out) {
  const int lane = threadIdx.x & 31;
  const int wid  = threadIdx.x >> 5;
  const int mo   = (blockIdx.x * 8 + wid) * 64;  // c_out tile base
  const int nc   = blockIdx.y * 32;              // pixel tile base
  const int b    = blockIdx.z;
  const int m    = lane & 15;
  const int kh   = lane >> 4;
  const __bf16* Ytb = Yt + (size_t)b * SP * CH;
  const v16bf zf = {};

  int py[2], px[2], pcol[2];
#pragma unroll
  for (int j = 0; j < 2; ++j) {
    pcol[j] = nc + 16 * j + m;
    py[j] = pcol[j] / 7;
    px[j] = pcol[j] % 7;
  }

  v8f acc[4][2] = {};
#pragma unroll
  for (int tap = 0; tap < 9; ++tap) {
    const int dy = tap / 3 - 1, dx = tap % 3 - 1;
    const __bf16* aptr[4];
#pragma unroll
    for (int i = 0; i < 4; ++i)
      aptr[i] = wt + ((size_t)tap * CH + (mo + 16 * i + m)) * CH;
    const __bf16* bptr[2];
    bool ok[2];
#pragma unroll
    for (int j = 0; j < 2; ++j) {
      int qy = py[j] + dy, qx = px[j] + dx;
      ok[j] = (pcol[j] < SP) && (qy >= 0) && (qy < 7) && (qx >= 0) && (qx < 7);
      int q = qy * 7 + qx;
      q = q < 0 ? 0 : (q >= SP ? SP - 1 : q);   // clamp, select below
      bptr[j] = Ytb + (size_t)q * CH;
    }
    for (int kb = 0; kb < CH; kb += 32) {
      v16bf af[4], bf[2];
#pragma unroll
      for (int j = 0; j < 2; ++j) {
        bf[j] = load_b_frag(bptr[j], kb, kh);
        bf[j] = ok[j] ? bf[j] : zf;             // cndmask, no branch
      }
#pragma unroll
      for (int i = 0; i < 4; ++i) af[i] = load_a_frag(aptr[i], kb, kh);
#pragma unroll
      for (int i = 0; i < 4; ++i)
#pragma unroll
        for (int j = 0; j < 2; ++j)
          acc[i][j] = __builtin_amdgcn_wmma_f32_16x16x32_bf16(
              false, af[i], false, bf[j], (short)0, acc[i][j], false, false);
    }
  }
  const int col0 = lane & 15;
  const int rb   = (lane >> 4) * 8;
#pragma unroll
  for (int i = 0; i < 4; ++i)
#pragma unroll
    for (int j = 0; j < 2; ++j) {
      int col = nc + 16 * j + col0;
      if (col < SP) {
#pragma unroll
        for (int r = 0; r < 8; ++r) {
          const int co = mo + 16 * i + rb + r;
          const size_t idx = ((size_t)b * CH + co) * SP + col;
          out[idx] = acc[i][j][r] + bias[co] + x[idx];
        }
      }
    }
}

extern "C" void kernel_launch(void* const* d_in, const int* in_sizes, int n_in,
                              void* d_out, int out_size, void* d_ws, size_t ws_size,
                              hipStream_t stream) {
  (void)in_sizes; (void)n_in; (void)out_size; (void)ws_size;
  const float* x      = (const float*)d_in[0];   // [B, C, 7, 7]
  const float* conv_w = (const float*)d_in[1];   // [C, C, 3, 3]
  const float* conv_b = (const float*)d_in[2];   // [C]
  float* out = (float*)d_out;                    // [B, C, 49]

  // Workspace (~121.5 MB):
  //   xp  bf16 [B][C][64]   8.0 MB   (K-padded rows: gram A & B)
  //   xt  bf16 [B][49][C]   6.4 MB   (transposed: mix B)
  //   wt  bf16 [9][C][C]   75.5 MB   (tap-major weights: conv A, L2-resident)
  //   sim f32  [C][C]      16.8 MB   (per-batch, reused)
  //   Wsm bf16 [C][C]       8.4 MB   (per-batch, reused)
  //   Yt  bf16 [B][49][C]   6.4 MB   (transposed Y: conv B)
  char* ws = (char*)d_ws;
  const size_t n_xp = (size_t)BATCH * CH * KPAD;
  const size_t n_xt = (size_t)BATCH * SP * CH;
  const size_t n_wt = (size_t)9 * CH * CH;
  __bf16* xp  = (__bf16*)ws;  ws += (n_xp * 2 + 255) & ~(size_t)255;
  __bf16* xt  = (__bf16*)ws;  ws += (n_xt * 2 + 255) & ~(size_t)255;
  __bf16* wt  = (__bf16*)ws;  ws += (n_wt * 2 + 255) & ~(size_t)255;
  float*  sim = (float*)ws;   ws += (size_t)CH * CH * 4;
  __bf16* Wsm = (__bf16*)ws;  ws += (size_t)CH * CH * 2;
  __bf16* Yt  = (__bf16*)ws;

  pad_xf  <<<(n_xp + 255) / 256, 256, 0, stream>>>(x, xp, (int)n_xp);
  trans_xf<<<(n_xt + 255) / 256, 256, 0, stream>>>(x, xt, (int)n_xt);
  trans_w <<<(n_wt + 255) / 256, 256, 0, stream>>>(conv_w, wt, (int)n_wt);

  for (int b = 0; b < BATCH; ++b) {
    gemm_gram<<<dim3(8, 64), 256, 0, stream>>>(xp + (size_t)b * CH * KPAD, sim);
    softmax_neg_rows<<<CH, 256, 0, stream>>>(sim, Wsm);
    gemm_mix<<<dim3(4, 2), 256, 0, stream>>>(Wsm, xt + (size_t)b * SP * CH,
                                             Yt + (size_t)b * SP * CH);
  }
  conv3x3_bias_res<<<dim3(4, 2, BATCH), 256, 0, stream>>>(Yt, wt, conv_b, x, out);
}
